// LFreqEmbed_87213605912677
// MI455X (gfx1250) — matrix operand order repaired and checked by reading, not compiled, not run on hardware
//
#include <hip/hip_runtime.h>
#include <hip/hip_bf16.h>

// ---- problem constants (from reference) ----
#define IMG   224
#define PLANE (IMG * IMG)          // 50176
#define CIN   3
#define K448  448                  // IMG * PS
#define DIN   1344                 // K448 * CIN
#define LTOK  112                  // token count L
#define EMB   768
#define NBATCH 256

// ---- tiling ----
#define BT 32                      // batch-rows per workgroup (M tile)
#define ET 64                      // embed cols per workgroup (N tile)
#define KC 32                      // K chunk (one bf16 WMMA deep)
#define NCHUNK (DIN / KC)          // 42
#define APITCH 40                  // bf16 elems per LDS row (80B, 16B aligned, cf-free)
#define WPITCH 40

typedef __attribute__((ext_vector_type(16))) __bf16 v16bf;
typedef __attribute__((ext_vector_type(8)))  __bf16 v8bf;
typedef __attribute__((ext_vector_type(8)))  float  v8f;
typedef __attribute__((ext_vector_type(4)))  float  v4f;

static __device__ inline v16bf cat8(v8bf a, v8bf b) {
  return __builtin_shufflevector(a, b, 0,1,2,3,4,5,6,7,8,9,10,11,12,13,14,15);
}

static __device__ inline void split_bf16(float v, __bf16* hi, __bf16* lo) {
  __bf16 h = (__bf16)v;            // RNE f32 -> bf16
  *hi = h;
  *lo = (__bf16)(v - (float)h);    // residual, captures ~8 more mantissa bits
}

__global__ __launch_bounds__(128)
void lfreq_embed_wmma_bf16x3(const float* __restrict__ x,
                             const float* __restrict__ W,
                             const float* __restrict__ bias,
                             const int*   __restrict__ a_idx,
                             const int*   __restrict__ b_idx,
                             float*       __restrict__ out) {
  __shared__ __align__(16) __bf16 Ahi[BT][APITCH];
  __shared__ __align__(16) __bf16 Alo[BT][APITCH];
  __shared__ __align__(16) __bf16 Whi[ET][WPITCH];
  __shared__ __align__(16) __bf16 Wlo[ET][WPITCH];

  const int tid = threadIdx.x;
  const int e0  = blockIdx.x * ET;       // embed tile base
  const int b0  = blockIdx.y * BT;       // batch tile base
  const int l   = blockIdx.z;            // token level (gather pattern fixed)

  const int wave = tid >> 5;
  const int lane = tid & 31;
  const int m0w  = (wave & 1) * 16;      // wave's 16-row M sub-tile
  const int nb   = (wave >> 1) * 32;     // wave's 32-col N sub-block
  const int h    = lane >> 4;            // lane half (WMMA K split)
  const int ln   = lane & 15;

  v8f acc0 = {};                          // N cols nb..nb+15
  v8f acc1 = {};                          // N cols nb+16..nb+31

  // per-thread geometry for the gather fill (32 rows x 32 k per chunk)
  const int krel  = tid & 31;             // k column this thread gathers
  const int brow0 = tid >> 5;             // base batch row (+4j)

  for (int ci = 0; ci < NCHUNK; ++ci) {
    const int d0  = ci * KC;
    const int cch = d0 / K448;            // input channel of this chunk
    const int k0  = d0 - cch * K448;

    // gather offset within one image plane: depends only on (l, k)
    const int ti  = l * K448 + k0 + krel;
    const int off = a_idx[ti] * IMG + b_idx[ti];
    const float* xc = x + cch * PLANE + off;

    __syncthreads();                      // previous chunk fully consumed

    // ---- A panel: gather 32 batch rows x 32 k, split into bf16 hi/lo ----
    #pragma unroll
    for (int j = 0; j < 8; ++j) {
      const int br = brow0 + 4 * j;
      const float v = xc[(b0 + br) * (CIN * PLANE)];
      __bf16 hi, lo;  split_bf16(v, &hi, &lo);
      Ahi[br][krel] = hi;
      Alo[br][krel] = lo;
    }

    // ---- W panel: 64 rows x 32 k, float4 coalesced, split hi/lo ----
    #pragma unroll
    for (int j = 0; j < 4; ++j) {
      const int q  = tid + 128 * j;       // 512 float4 total
      const int er = q >> 3;
      const int c4 = (q & 7) * 4;
      const v4f wv = *(const v4f*)&W[(e0 + er) * DIN + d0 + c4];
      #pragma unroll
      for (int u = 0; u < 4; ++u) {
        __bf16 hi, lo;  split_bf16(wv[u], &hi, &lo);
        Whi[er][c4 + u] = hi;
        Wlo[er][c4 + u] = lo;
      }
    }
    __syncthreads();

    // ---- fragments: A 16x32 bf16 layout (K = 16h + 0..15) ----
    const int am = m0w + ln;
    const v16bf ahi = cat8(*(const v8bf*)&Ahi[am][16 * h],
                           *(const v8bf*)&Ahi[am][16 * h + 8]);
    const v16bf alo = cat8(*(const v8bf*)&Alo[am][16 * h],
                           *(const v8bf*)&Alo[am][16 * h + 8]);

    // ---- B fragments: 32x16 bf16 layout (K = 16h + 0..15), n = col ----
    const int n0 = nb + ln;
    const int n1 = nb + 16 + ln;
    const v16bf bhi0 = cat8(*(const v8bf*)&Whi[n0][16 * h],
                            *(const v8bf*)&Whi[n0][16 * h + 8]);
    const v16bf blo0 = cat8(*(const v8bf*)&Wlo[n0][16 * h],
                            *(const v8bf*)&Wlo[n0][16 * h + 8]);
    const v16bf bhi1 = cat8(*(const v8bf*)&Whi[n1][16 * h],
                            *(const v8bf*)&Whi[n1][16 * h + 8]);
    const v16bf blo1 = cat8(*(const v8bf*)&Wlo[n1][16 * h],
                            *(const v8bf*)&Wlo[n1][16 * h + 8]);

    // ---- bf16x3: hi*hi + hi*lo + lo*hi (lo*lo dropped, ~2^-16) ----
    acc0 = __builtin_amdgcn_wmma_f32_16x16x32_bf16(false, ahi, false, bhi0,
                                                   (short)0, acc0, false, false);
    acc0 = __builtin_amdgcn_wmma_f32_16x16x32_bf16(false, ahi, false, blo0,
                                                   (short)0, acc0, false, false);
    acc0 = __builtin_amdgcn_wmma_f32_16x16x32_bf16(false, alo, false, bhi0,
                                                   (short)0, acc0, false, false);
    acc1 = __builtin_amdgcn_wmma_f32_16x16x32_bf16(false, ahi, false, bhi1,
                                                   (short)0, acc1, false, false);
    acc1 = __builtin_amdgcn_wmma_f32_16x16x32_bf16(false, ahi, false, blo1,
                                                   (short)0, acc1, false, false);
    acc1 = __builtin_amdgcn_wmma_f32_16x16x32_bf16(false, alo, false, bhi1,
                                                   (short)0, acc1, false, false);
  }

  // ---- epilogue: bias add + store (C/D layout: M = r + 8h, N = ln) ----
  const float bz0 = bias[e0 + nb + ln];
  const float bz1 = bias[e0 + nb + 16 + ln];
  #pragma unroll
  for (int r = 0; r < 8; ++r) {
    const int m = m0w + r + 8 * h;                 // row within BT tile
    const long row = (long)(b0 + m) * LTOK + l;    // output row = b*L + l
    out[row * EMB + e0 + nb + ln]      = acc0[r] + bz0;
    out[row * EMB + e0 + nb + 16 + ln] = acc1[r] + bz1;
  }
}

extern "C" void kernel_launch(void* const* d_in, const int* in_sizes, int n_in,
                              void* d_out, int out_size, void* d_ws, size_t ws_size,
                              hipStream_t stream) {
  const float* x    = (const float*)d_in[0];
  const float* W    = (const float*)d_in[1];
  const float* bias = (const float*)d_in[2];
  const int*   ai   = (const int*)d_in[3];
  const int*   bi   = (const int*)d_in[4];
  float* out = (float*)d_out;

  dim3 grid(EMB / ET, NBATCH / BT, LTOK);  // (12, 8, 112)
  dim3 block(128);                         // 4 waves (wave32)
  hipLaunchKernelGGL(lfreq_embed_wmma_bf16x3, grid, block, 0, stream,
                     x, W, bias, ai, bi, out);
}
